// OrthogonalMultiheadAttention_34093450395958
// MI455X (gfx1250) — matrix operand order repaired and checked by reading, not compiled, not run on hardware
//
#include <hip/hip_runtime.h>
#include <hip/hip_bf16.h>
#include <math.h>

// ---------------------------------------------------------------------------
// OrthogonalMultiheadAttention for MI455X (gfx1250, wave32, WMMA)
//
// All GEMMs run through v_wmma_f32_16x16x32_bf16 (bf16 inputs, f32 accum).
// Block = 256 threads = 8 waves; block tile 128x128, K-step 32.
// Wave grid 4(M) x 2(N); wave tile 32x64 -> 2x4 = 8 wmma accumulators (v8f).
// A and B are staged f32->bf16 into LDS (padded stride 40 halves = 80B to
// avoid bank conflicts); B is stored K-transposed so B-fragments are
// contiguous ds_load_b128 pairs per the ISA 7.12.2 layout. f32->bf16 uses
// packed v_cvt_pk_bf16_f32 when available, with ds_store_b128 staging.
// ---------------------------------------------------------------------------

typedef __attribute__((ext_vector_type(16))) __bf16        v16bf;
typedef __attribute__((ext_vector_type(8)))  float         v8f;
typedef __attribute__((ext_vector_type(4)))  unsigned int  u32x4;
typedef __attribute__((ext_vector_type(8)))  unsigned int  u32x8;

#define EMBED 1024
#define NHEAD 16
#define HDIM  64
#define BATCH 2
#define SEQ   2048

__device__ __forceinline__ unsigned short f2bf(float f) {
  // round-to-nearest-even f32 -> bf16 (fallback path)
  unsigned int u = __float_as_uint(f);
  u += 0x7fffu + ((u >> 16) & 1u);
  return (unsigned short)(u >> 16);
}

// Packed conversion: returns {bf16(a) in [15:0], bf16(b) in [31:16]}
__device__ __forceinline__ unsigned int pk2bf(float a, float b) {
#if __has_builtin(__builtin_amdgcn_cvt_pk_bf16_f32)
  auto pk = __builtin_amdgcn_cvt_pk_bf16_f32(a, b);
  return __builtin_bit_cast(unsigned int, pk);
#else
  return (unsigned int)f2bf(a) | ((unsigned int)f2bf(b) << 16);
#endif
}

union ABFrag { u32x8 u; v16bf b; };

// C[M,N] = alpha * (A @ B) + beta * D
//  TA: A stored K-major (A[m,k] = Ap[k*lda + m])  else Ap[m*lda + k]
//  TB: B stored N-major (B[k,n] = Bp[n*ldb + k])  else Bp[k*ldb + n]
// Requires M % 128 == 0, K % 32 == 0. N may be ragged (guarded).
template<bool TA, bool TB>
__global__ __launch_bounds__(256) void gemm_bf16_wmma(
    const float* __restrict__ A, const float* __restrict__ Bm,
    const float* __restrict__ D, float* __restrict__ C,
    int M, int N, int K, int lda, int ldb, int ldd, int ldc,
    float alpha, float beta)
{
  constexpr int BM = 128, BN = 128, BK = 32, LDS_K = BK + 8;
  __shared__ __align__(16) unsigned short As[BM][LDS_K];
  __shared__ __align__(16) unsigned short Bs[BN][LDS_K];  // [n][k] transposed

  const int t      = threadIdx.x;
  const int lane   = t & 31;
  const int wid    = t >> 5;
  const int waveM  = (wid & 3) * 32;   // 4 waves along M
  const int waveN  = (wid >> 2) * 64;  // 2 waves along N
  const int blockN = blockIdx.x * BN;
  const int blockM = blockIdx.y * BM;
  const int lhalf  = (lane >= 16) ? 1 : 0;
  const int l15    = lane & 15;

  v8f acc[2][4];
#pragma unroll
  for (int im = 0; im < 2; ++im)
#pragma unroll
    for (int in = 0; in < 4; ++in)
      acc[im][in] = v8f{0.f, 0.f, 0.f, 0.f, 0.f, 0.f, 0.f, 0.f};

  for (int kb = 0; kb < K; kb += BK) {
    // ---------------- stage A tile (128 x 32) ----------------
    if (!TA) {
      const int m = t >> 1, k0 = (t & 1) * 16;
      const float* src = A + (size_t)(blockM + m) * lda + kb + k0;
      unsigned int* dst = (unsigned int*)&As[m][k0];  // 16B aligned
#pragma unroll
      for (int i = 0; i < 16; i += 8) {
        float4 f0 = *(const float4*)(src + i);
        float4 f1 = *(const float4*)(src + i + 4);
        u32x4 p = {pk2bf(f0.x, f0.y), pk2bf(f0.z, f0.w),
                   pk2bf(f1.x, f1.y), pk2bf(f1.z, f1.w)};
        *(u32x4*)(dst + (i >> 1)) = p;                // ds_store_b128
      }
      if (kb + BK < K) __builtin_prefetch(src + BK, 0, 1);
    } else {
      const int kk = t >> 3, m0 = (t & 7) * 16;
      const float* src = A + (size_t)(kb + kk) * lda + blockM + m0;
#pragma unroll
      for (int i = 0; i < 16; i += 4) {
        float4 f = *(const float4*)(src + i);
        As[m0 + i + 0][kk] = f2bf(f.x); As[m0 + i + 1][kk] = f2bf(f.y);
        As[m0 + i + 2][kk] = f2bf(f.z); As[m0 + i + 3][kk] = f2bf(f.w);
      }
      if (kb + BK < K) __builtin_prefetch(src + (size_t)BK * lda, 0, 1);
    }
    // ---------------- stage B tile, transposed to [n][k] ----------------
    if (TB) {
      const int n = t >> 1, k0 = (t & 1) * 16;
      unsigned int* dst = (unsigned int*)&Bs[n][k0];  // 16B aligned
      const int gn = blockN + n;
      if (gn < N) {
        const float* src = Bm + (size_t)gn * ldb + kb + k0;
#pragma unroll
        for (int i = 0; i < 16; i += 8) {
          float4 f0 = *(const float4*)(src + i);
          float4 f1 = *(const float4*)(src + i + 4);
          u32x4 p = {pk2bf(f0.x, f0.y), pk2bf(f0.z, f0.w),
                     pk2bf(f1.x, f1.y), pk2bf(f1.z, f1.w)};
          *(u32x4*)(dst + (i >> 1)) = p;              // ds_store_b128
        }
        if (kb + BK < K) __builtin_prefetch(src + BK, 0, 1);
      } else {
        u32x4 z = {0u, 0u, 0u, 0u};
        *(u32x4*)(dst + 0) = z;
        *(u32x4*)(dst + 4) = z;
      }
    } else {
      const int kk = t >> 3, n0 = (t & 7) * 16;
      const float* src = Bm + (size_t)(kb + kk) * ldb + blockN + n0;
#pragma unroll
      for (int i = 0; i < 16; ++i) {
        const int gn = blockN + n0 + i;
        float f = (gn < N) ? src[i] : 0.0f;
        Bs[n0 + i][kk] = f2bf(f);
      }
    }
    __syncthreads();

    // ---------------- fragments (ISA 7.12.2 layouts) ----------------
    // A 16x32 bf16: lanes<16 hold K[0..7]+K[16..23]; lanes>=16 K[8..15]+K[24..31]
    ABFrag a[2];
#pragma unroll
    for (int im = 0; im < 2; ++im) {
      const unsigned short* p = &As[waveM + im * 16 + l15][lhalf * 8];
      u32x4 lo = *(const u32x4*)p;          // K = h .. h+7
      u32x4 hi = *(const u32x4*)(p + 16);   // K = h+16 .. h+23
      a[im].u = u32x8{lo.x, lo.y, lo.z, lo.w, hi.x, hi.y, hi.z, hi.w};
    }
    // B 32x16 bf16: lanes<16 hold K[0..7]+K[16..23] of col N=lane; lanes>=16 K[8..15]+K[24..31]
    ABFrag b[4];
#pragma unroll
    for (int in = 0; in < 4; ++in) {
      const unsigned short* p = &Bs[waveN + in * 16 + l15][lhalf * 8];
      u32x4 lo = *(const u32x4*)p;          // K = h .. h+7
      u32x4 hi = *(const u32x4*)(p + 16);   // K = h+16 .. h+23
      b[in].u = u32x8{lo.x, lo.y, lo.z, lo.w, hi.x, hi.y, hi.z, hi.w};
    }
#pragma unroll
    for (int im = 0; im < 2; ++im)
#pragma unroll
      for (int in = 0; in < 4; ++in)
        acc[im][in] = __builtin_amdgcn_wmma_f32_16x16x32_bf16(
            false, a[im].b, false, b[in].b, (short)0, acc[im][in], false, false);
    __syncthreads();
  }

  // ---------------- epilogue: C = alpha*acc + beta*D ----------------
  // C/D layout per 16x16 f32 tile: VGPR r, lanes<16 -> M=r, lanes>=16 -> M=8+r
#pragma unroll
  for (int im = 0; im < 2; ++im) {
#pragma unroll
    for (int in = 0; in < 4; ++in) {
      const int col = blockN + waveN + in * 16 + l15;
      if (col >= N) continue;
      const int row0 = blockM + waveM + im * 16 + lhalf * 8;
#pragma unroll
      for (int r = 0; r < 8; ++r) {
        float val = alpha * acc[im][in][r];
        if (beta != 0.0f)
          val = fmaf(beta, D[(size_t)(row0 + r) * ldd + col], val);
        C[(size_t)(row0 + r) * ldc + col] = val;
      }
    }
  }
}

// Row softmax, ncols must be 2048 (= 8 * 256). One block per row.
__global__ __launch_bounds__(256) void softmax_rows(float* __restrict__ S, int ncols)
{
  __shared__ float red[256];
  float* row = S + (size_t)blockIdx.x * ncols;
  const int t = threadIdx.x;

  float vals[8];
  float vmax = -INFINITY;
#pragma unroll
  for (int i = 0; i < 8; ++i) {
    vals[i] = row[t + i * 256];
    vmax = fmaxf(vmax, vals[i]);
  }
  red[t] = vmax; __syncthreads();
  for (int s = 128; s > 0; s >>= 1) {
    if (t < s) red[t] = fmaxf(red[t], red[t + s]);
    __syncthreads();
  }
  vmax = red[0]; __syncthreads();

  float sum = 0.f;
#pragma unroll
  for (int i = 0; i < 8; ++i) {
    vals[i] = __expf(vals[i] - vmax);
    sum += vals[i];
  }
  red[t] = sum; __syncthreads();
  for (int s = 128; s > 0; s >>= 1) {
    if (t < s) red[t] += red[t + s];
    __syncthreads();
  }
  const float inv = 1.0f / red[0];
#pragma unroll
  for (int i = 0; i < 8; ++i) row[t + i * 256] = vals[i] * inv;
}

// ---------------------------------------------------------------------------

template<bool TA, bool TB>
static void gemm(hipStream_t s, const float* A, const float* B, const float* D,
                 float* C, int M, int N, int K, int lda, int ldb, int ldd,
                 int ldc, float alpha, float beta)
{
  dim3 grid((N + 127) / 128, (M + 127) / 128), blk(256);
  gemm_bf16_wmma<TA, TB><<<grid, blk, 0, s>>>(A, B, D, C, M, N, K,
                                              lda, ldb, ldd, ldc, alpha, beta);
}

extern "C" void kernel_launch(void* const* d_in, const int* in_sizes, int n_in,
                              void* d_out, int out_size, void* d_ws, size_t ws_size,
                              hipStream_t stream) {
  (void)in_sizes; (void)n_in; (void)out_size; (void)ws_size;
  const float* query = (const float*)d_in[0];
  const float* key   = (const float*)d_in[1];
  const float* value = (const float*)d_in[2];
  const float* Wq    = (const float*)d_in[3];
  const float* Wk    = (const float*)d_in[4];
  const float* Wv    = (const float*)d_in[5];
  const float* Wo    = (const float*)d_in[6];
  float* out = (float*)d_out;

  const int E = EMBED, Sq = SEQ, BS = BATCH * SEQ;
  constexpr size_t M1 = (size_t)EMBED * EMBED;        // 1M floats
  constexpr size_t M4 = (size_t)BATCH * SEQ * EMBED;  // 4M floats
  float* ws   = (float*)d_ws;                         // ~96 MB total
  float* Gk   = ws;                                   // Wk^T Wk
  float* Gq   = ws + M1;                              // Wq^T Wq
  float* Wqo  = ws + 2 * M1;
  float* Wko  = ws + 3 * M1;
  float* qb   = ws + 4 * M1;                          // [BS, E]
  float* kb   = qb + M4;
  float* vb   = kb + M4;
  float* ao   = vb + M4;                              // attention output [BS, E]
  float* sc   = ao + M4;                              // scores [S, S], reused per (b,h)

  // 1) Gram matrices: G = W^T @ W
  gemm<true, false>(stream, Wk, Wk, nullptr, Gk, E, E, E, E, E, 0, E, 1.f, 0.f);
  gemm<true, false>(stream, Wq, Wq, nullptr, Gq, E, E, E, E, E, 0, E, 1.f, 0.f);

  // 2) Orthogonalize (both from ORIGINAL weights): W_o = W - W @ G
  gemm<false, false>(stream, Wq, Gk, Wq, Wqo, E, E, E, E, E, E, E, -1.f, 1.f);
  gemm<false, false>(stream, Wk, Gq, Wk, Wko, E, E, E, E, E, E, E, -1.f, 1.f);

  // 3) Projections: x @ W^T
  gemm<false, true>(stream, query, Wqo, nullptr, qb, BS, E, E, E, E, 0, E, 1.f, 0.f);
  gemm<false, true>(stream, key,   Wko, nullptr, kb, BS, E, E, E, E, 0, E, 1.f, 0.f);
  gemm<false, true>(stream, value, Wv,  nullptr, vb, BS, E, E, E, E, 0, E, 1.f, 0.f);

  // 4) Attention per (batch, head); scores buffer recycled on the stream
  const float scale = 1.0f / 8.0f;  // 1/sqrt(64)
  for (int b = 0; b < BATCH; ++b) {
    for (int h = 0; h < NHEAD; ++h) {
      const size_t off = (size_t)b * SEQ * E + (size_t)h * HDIM;
      const float* Qh = qb + off;
      const float* Kh = kb + off;
      const float* Vh = vb + off;
      float*       Oh = ao + off;
      // scores = scale * Qh @ Kh^T   [S, S], K = 64
      gemm<false, true>(stream, Qh, Kh, nullptr, sc, Sq, Sq, HDIM, E, E, 0, Sq,
                        scale, 0.f);
      softmax_rows<<<Sq, 256, 0, stream>>>(sc, Sq);
      // Oh = softmax(scores) @ Vh    [S, 64], strided into [BS, E]
      gemm<false, false>(stream, sc, Vh, nullptr, Oh, Sq, HDIM, Sq, Sq, E, 0, E,
                         1.f, 0.f);
    }
  }

  // 5) Output projection: out = ao @ Wo^T
  gemm<false, true>(stream, ao, Wo, nullptr, out, BS, E, E, E, E, 0, E, 1.f, 0.f);
}
